// GraphClassifier_44152263803373
// MI455X (gfx1250) — compile-verified, matrix operand order checked by reading, not run on hardware
//
#include <hip/hip_runtime.h>
#include <math.h>

typedef __attribute__((ext_vector_type(16))) _Float16 v16h;
typedef __attribute__((ext_vector_type(4)))  _Float16 v4h;
typedef __attribute__((ext_vector_type(2)))  _Float16 v2h;
typedef __attribute__((ext_vector_type(8)))  float    v8f;

#define DIM 64
#define NUM_GRAPHS 64

__device__ __forceinline__ float lrelu(float t) { return t > 0.f ? t : 0.2f * t; }

// Order-preserving float -> uint key so we can do segment-max with atomicMax(uint).
__device__ __forceinline__ unsigned fkey(float f) {
    unsigned u = __float_as_uint(f);
    return (u & 0x80000000u) ? ~u : (u | 0x80000000u);
}
__device__ __forceinline__ float fdecode(unsigned k) {
    unsigned u = (k & 0x80000000u) ? (k ^ 0x80000000u) : ~k;
    return __uint_as_float(u);
}

// ---------------------------------------------------------------------------
// xl = x @ Wl^T + bl ; xr = x @ Wr^T + br   (per layer), via WMMA f16->f32.
// Block: 256 threads = 8 waves; each wave owns a 16-row tile; weights in LDS.
// Outputs stored as f16 to halve the irregular edge-gather traffic downstream.
// ---------------------------------------------------------------------------
__global__ void __launch_bounds__(256)
transform_kernel(const float* __restrict__ x,
                 const float* __restrict__ Wl, const float* __restrict__ bl,
                 const float* __restrict__ Wr, const float* __restrict__ br,
                 _Float16* __restrict__ xl, _Float16* __restrict__ xr, int N)
{
    __shared__ _Float16 WlS[DIM * DIM];
    __shared__ _Float16 WrS[DIM * DIM];
    for (int i = threadIdx.x; i < DIM * DIM; i += 256) {
        WlS[i] = (_Float16)Wl[i];
        WrS[i] = (_Float16)Wr[i];
    }
    __syncthreads();

    const int lane = threadIdx.x & 31;   // wave32
    const int wave = threadIdx.x >> 5;
    const int col  = lane & 15;
    const int hi   = lane >> 4;
    const int base = blockIdx.x * 128 + wave * 16;

    v8f accl[4] = {};
    v8f accr[4] = {};

    int mrow = base + col;               // A-matrix row held by this lane
    if (mrow >= N) mrow = N - 1;         // branchless clamp: EXEC stays all-1s

    #pragma unroll
    for (int kc = 0; kc < 2; ++kc) {
        const int K0 = kc * 32;
        v16h a;
        #pragma unroll
        for (int v = 0; v < 8; ++v) {
            const int k = K0 + ((v < 4) ? (hi * 8 + 2 * v)
                                        : (16 + hi * 8 + 2 * (v - 4)));
            const float2 xv = *(const float2*)(x + (size_t)mrow * DIM + k);
            a[2 * v]     = (_Float16)xv.x;
            a[2 * v + 1] = (_Float16)xv.y;
        }
        #pragma unroll
        for (int t = 0; t < 4; ++t) {
            const int n = 16 * t + col;  // B-matrix column held by this lane
            v16h bf, cf;
            #pragma unroll
            for (int v = 0; v < 8; ++v) {
                const int k = K0 + ((v < 4) ? (hi * 8 + 2 * v)
                                            : (16 + hi * 8 + 2 * (v - 4)));
                const v2h w2l = *(const v2h*)&WlS[n * DIM + k];
                const v2h w2r = *(const v2h*)&WrS[n * DIM + k];
                bf[2 * v] = w2l[0]; bf[2 * v + 1] = w2l[1];
                cf[2 * v] = w2r[0]; cf[2 * v + 1] = w2r[1];
            }
            accl[t] = __builtin_amdgcn_wmma_f32_16x16x32_f16(
                false, a, false, bf, (short)0, accl[t], false, false);
            accr[t] = __builtin_amdgcn_wmma_f32_16x16x32_f16(
                false, a, false, cf, (short)0, accr[t], false, false);
        }
    }

    // C/D layout: lane col = N index, VGPR v -> M = v + 8*hi
    #pragma unroll
    for (int t = 0; t < 4; ++t) {
        const int n   = 16 * t + col;
        const float bvl = bl[n];
        const float bvr = br[n];
        #pragma unroll
        for (int v = 0; v < 8; ++v) {
            const int row = base + 8 * hi + v;
            if (row < N) {
                xl[(size_t)row * DIM + n] = (_Float16)(accl[t][v] + bvl);
                xr[(size_t)row * DIM + n] = (_Float16)(accr[t][v] + bvr);
            }
        }
    }
}

// ---------------------------------------------------------------------------
// e = dot(leaky_relu(xl[src] + xr[dst], 0.2), att); segment-max into mkey[dst]
// 2 edges per wave, 16 lanes x 4 f16 features (8B/lane) per edge.
// ---------------------------------------------------------------------------
__global__ void __launch_bounds__(256)
edge_score_kernel(const _Float16* __restrict__ xl, const _Float16* __restrict__ xr,
                  const float* __restrict__ att, const int* __restrict__ ei,
                  long long E, long long Etot,
                  float* __restrict__ e_out, unsigned* __restrict__ mkey)
{
    const int lane = threadIdx.x & 31;
    const int wave = threadIdx.x >> 5;
    const int col  = lane & 15;
    const long long edge = (((long long)blockIdx.x * 8 + wave) << 1) + (lane >> 4);
    if (edge >= Etot) return;

    int s, d;
    if (edge < E) { s = ei[edge]; d = ei[E + edge]; }
    else          { s = d = (int)(edge - E); }

    const v4h xa = *(const v4h*)(xl + (size_t)s * DIM + col * 4);
    const v4h xb = *(const v4h*)(xr + (size_t)d * DIM + col * 4);
    const float4 w = *(const float4*)(att + col * 4);

    float p = lrelu((float)xa[0] + (float)xb[0]) * w.x
            + lrelu((float)xa[1] + (float)xb[1]) * w.y
            + lrelu((float)xa[2] + (float)xb[2]) * w.z
            + lrelu((float)xa[3] + (float)xb[3]) * w.w;
    #pragma unroll
    for (int m = 8; m >= 1; m >>= 1) p += __shfl_xor(p, m, 16);

    if (col == 0) {
        e_out[edge] = p;
        atomicMax(&mkey[d], fkey(p));
    }
}

__global__ void decode_m_kernel(const unsigned* __restrict__ mkey,
                                float* __restrict__ m, int N)
{
    const int i = blockIdx.x * 256 + threadIdx.x;
    if (i < N) {
        const unsigned k = mkey[i];
        m[i] = (k == 0u) ? 0.f : fdecode(k);  // all nodes have self-loops
    }
}

// num = exp(e - m[dst]); den[dst] += num   (num overwrites e in place)
__global__ void edge_exp_kernel(const int* __restrict__ ei,
                                long long E, long long Etot,
                                const float* __restrict__ m,
                                float* __restrict__ e_num,
                                float* __restrict__ den)
{
    const long long i = (long long)blockIdx.x * 256 + threadIdx.x;
    if (i >= Etot) return;
    const int d = (i < E) ? ei[E + i] : (int)(i - E);
    const float v = __expf(e_num[i] - m[d]);
    e_num[i] = v;
    atomicAdd(&den[d], v);
}

// out[dst] += alpha * xl[src]; alpha = num / max(den[dst], 1e-16)
__global__ void __launch_bounds__(256)
aggregate_kernel(const _Float16* __restrict__ xl, const float* __restrict__ num,
                 const float* __restrict__ den, const int* __restrict__ ei,
                 long long E, long long Etot, float* __restrict__ out)
{
    const int lane = threadIdx.x & 31;
    const int wave = threadIdx.x >> 5;
    const int col  = lane & 15;
    const long long edge = (((long long)blockIdx.x * 8 + wave) << 1) + (lane >> 4);
    if (edge >= Etot) return;

    int s, d;
    if (edge < E) { s = ei[edge]; d = ei[E + edge]; }
    else          { s = d = (int)(edge - E); }

    const float alpha = num[edge] / fmaxf(den[d], 1e-16f);
    const v4h xa = *(const v4h*)(xl + (size_t)s * DIM + col * 4);
    float* o = out + (size_t)d * DIM + col * 4;
    atomicAdd(o + 0, alpha * (float)xa[0]);
    atomicAdd(o + 1, alpha * (float)xa[1]);
    atomicAdd(o + 2, alpha * (float)xa[2]);
    atomicAdd(o + 3, alpha * (float)xa[3]);
}

__global__ void bias_relu_kernel(const float* __restrict__ agg,
                                 const float* __restrict__ bias,
                                 float* __restrict__ xo, long long total)
{
    const long long i = (long long)blockIdx.x * 256 + threadIdx.x;
    if (i < total) {
        const int f = (int)(i & (DIM - 1));
        xo[i] = fmaxf(agg[i] + bias[f], 0.f);
    }
}

__global__ void pool_sum_kernel(const float* __restrict__ x,
                                const int* __restrict__ batch,
                                float* __restrict__ pooled, long long total)
{
    const long long i = (long long)blockIdx.x * 256 + threadIdx.x;
    if (i < total) {
        const long long n = i >> 6;
        const int f = (int)(i & (DIM - 1));
        atomicAdd(&pooled[(size_t)batch[n] * DIM + f], x[i]);
    }
}

__global__ void pool_cnt_kernel(const int* __restrict__ batch,
                                float* __restrict__ cnt, int N)
{
    const int i = blockIdx.x * 256 + threadIdx.x;
    if (i < N) atomicAdd(&cnt[batch[i]], 1.f);
}

__global__ void final_kernel(const float* __restrict__ pooled,
                             const float* __restrict__ cnt,
                             const float* __restrict__ W,
                             const float* __restrict__ b,
                             float* __restrict__ out, int C)
{
    const int g = blockIdx.x;
    const int c = threadIdx.x;
    if (c >= C) return;
    const float inv = 1.f / fmaxf(cnt[g], 1.f);
    float acc = b[c];
    for (int d = 0; d < DIM; ++d)
        acc += pooled[g * DIM + d] * inv * W[c * DIM + d];
    out[g * C + c] = acc;
}

extern "C" void kernel_launch(void* const* d_in, const int* in_sizes, int n_in,
                              void* d_out, int out_size, void* d_ws, size_t ws_size,
                              hipStream_t stream)
{
    (void)n_in; (void)out_size; (void)ws_size;
    const float* x0    = (const float*)d_in[0];
    const int*   ei    = (const int*)d_in[1];
    const int*   batch = (const int*)d_in[2];
    const float* Wl    = (const float*)d_in[3];
    const float* bl    = (const float*)d_in[4];
    const float* Wr    = (const float*)d_in[5];
    const float* br    = (const float*)d_in[6];
    const float* att   = (const float*)d_in[7];
    const float* gb    = (const float*)d_in[8];
    const float* linW  = (const float*)d_in[9];
    const float* linb  = (const float*)d_in[10];
    float* out = (float*)d_out;

    const long long N    = in_sizes[0] / DIM;
    const long long E    = in_sizes[1] / 2;
    const long long Etot = E + N;
    const int L = in_sizes[3] / (DIM * DIM);
    const int C = in_sizes[9] / DIM;

    char* p = (char*)d_ws;
    auto carve = [&](size_t bytes) {
        char* q = p;
        p += (bytes + 255) & ~(size_t)255;
        return (void*)q;
    };
    _Float16* xl     = (_Float16*)carve((size_t)N * DIM * 2);
    _Float16* xr     = (_Float16*)carve((size_t)N * DIM * 2);
    float*    agg    = (float*)carve((size_t)N * DIM * 4);
    float*    xbA    = (float*)carve((size_t)N * DIM * 4);
    float*    xbB    = (float*)carve((size_t)N * DIM * 4);
    float*    e_num  = (float*)carve((size_t)Etot * 4);
    unsigned* mkey   = (unsigned*)carve((size_t)N * 4);
    float*    mval   = (float*)carve((size_t)N * 4);
    float*    den    = (float*)carve((size_t)N * 4);
    float*    pooled = (float*)carve((size_t)NUM_GRAPHS * DIM * 4);
    float*    cnt    = (float*)carve((size_t)NUM_GRAPHS * 4);

    const unsigned nodeBlocks = (unsigned)((N + 127) / 128);
    const unsigned edgeBlocks = (unsigned)((Etot + 15) / 16);   // 16 edges/block
    const unsigned eThrBlocks = (unsigned)((Etot + 255) / 256);
    const unsigned nThrBlocks = (unsigned)((N + 255) / 256);
    const unsigned ndBlocks   = (unsigned)((N * DIM + 255) / 256);

    const float* x = x0;
    for (int l = 0; l < L; ++l) {
        transform_kernel<<<nodeBlocks, 256, 0, stream>>>(
            x, Wl + (size_t)l * DIM * DIM, bl + l * DIM,
            Wr + (size_t)l * DIM * DIM, br + l * DIM, xl, xr, (int)N);
        hipMemsetAsync(mkey, 0, (size_t)N * 4, stream);
        hipMemsetAsync(den,  0, (size_t)N * 4, stream);
        hipMemsetAsync(agg,  0, (size_t)N * DIM * 4, stream);
        edge_score_kernel<<<edgeBlocks, 256, 0, stream>>>(
            xl, xr, att + l * DIM, ei, E, Etot, e_num, mkey);
        decode_m_kernel<<<nThrBlocks, 256, 0, stream>>>(mkey, mval, (int)N);
        edge_exp_kernel<<<eThrBlocks, 256, 0, stream>>>(ei, E, Etot, mval, e_num, den);
        aggregate_kernel<<<edgeBlocks, 256, 0, stream>>>(xl, e_num, den, ei, E, Etot, agg);
        float* xnext = (l & 1) ? xbB : xbA;
        bias_relu_kernel<<<ndBlocks, 256, 0, stream>>>(agg, gb + l * DIM, xnext, N * DIM);
        x = xnext;
    }

    hipMemsetAsync(pooled, 0, (size_t)NUM_GRAPHS * DIM * 4, stream);
    hipMemsetAsync(cnt,    0, (size_t)NUM_GRAPHS * 4, stream);
    pool_sum_kernel<<<ndBlocks, 256, 0, stream>>>(x, batch, pooled, N * DIM);
    pool_cnt_kernel<<<nThrBlocks, 256, 0, stream>>>(batch, cnt, (int)N);
    final_kernel<<<NUM_GRAPHS, 32, 0, stream>>>(pooled, cnt, linW, linb, out, C);
}
